// GATLinkPredictor_21131239096609
// MI455X (gfx1250) — matrix operand order repaired and checked by reading, not compiled, not run on hardware
//
#include <hip/hip_runtime.h>
#include <hip/hip_bf16.h>

typedef __attribute__((ext_vector_type(16))) _Float16 v16h;
typedef __attribute__((ext_vector_type(8)))  float    v8f;

#define NEG_SLOPE 0.2f

// ---------------------------------------------------------------------------
// Pack B[K,N] (f32, row-major) into WMMA B-lane layout as f16.
// Layout: Bp[(tn*KS + kk)*32 + lane] = v16h of B[kB+j+(j>=8?8:0)][tn*16+r],
// where kB = kk*32 + 8*hw  (hw = lane>>4, r = lane&15), j = 0..15.
// ---------------------------------------------------------------------------
__global__ void gat_pack_b(const float* __restrict__ B, v16h* __restrict__ Bp,
                           int N, int K)
{
  const int KS = K / 32;
  const int total = (N / 16) * KS * 32;
  const int i = blockIdx.x * blockDim.x + threadIdx.x;
  if (i >= total) return;
  const int lane = i & 31;
  const int t    = i >> 5;
  const int kk   = t % KS;
  const int tn   = t / KS;
  const int hw   = lane >> 4, r = lane & 15;
  const int col  = tn * 16 + r;
  const int kB   = kk * 32 + 8 * hw;
  v16h b;
  for (int j = 0; j < 16; ++j)
    b[j] = (_Float16)B[(size_t)(kB + j + (j >= 8 ? 8 : 0)) * N + col];
  Bp[(size_t)(tn * KS + kk) * 32 + lane] = b;
}

// ---------------------------------------------------------------------------
// WMMA GEMM: C[M,NN] = A[M,K] @ B[K,NN]; f32 A converted to f16 on the fly,
// B pre-packed f16, f32 accumulate. One wave per 16x16 tile, 8 waves/block.
// ---------------------------------------------------------------------------
template <int K, int NN>
__global__ __launch_bounds__(256) void gat_wmma_gemm(
    const float* __restrict__ A, const v16h* __restrict__ Bp,
    float* __restrict__ C, int M)
{
  constexpr int KS     = K / 32;
  constexpr int tilesN = NN / 16;
  const int tilesM = (M + 15) >> 4;
  const int tile   = blockIdx.x * (blockDim.x >> 5) + (threadIdx.x >> 5);
  if (tile >= tilesM * tilesN) return;
  const int tm   = tile / tilesN;
  const int tn   = tile - tm * tilesN;
  const int lane = threadIdx.x & 31;
  const int hw   = lane >> 4;
  const int r    = lane & 15;

  int arow = tm * 16 + r;
  if (arow >= M) arow = M - 1;            // clamp loads; stores guarded below
  const float* __restrict__ ap = A + (size_t)arow * K;
  const v16h*  __restrict__ bp = Bp + (size_t)tn * KS * 32 + lane;

  v8f acc = {};
#pragma unroll
  for (int kk = 0; kk < KS; ++kk) {
    const int k0  = kk * 32;
    const int kA0 = k0 + 8 * hw;          // elements 0..7
    const int kA1 = k0 + 16 + 8 * hw;     // elements 8..15
    float4 a0 = *(const float4*)(ap + kA0);
    float4 a1 = *(const float4*)(ap + kA0 + 4);
    float4 a2 = *(const float4*)(ap + kA1);
    float4 a3 = *(const float4*)(ap + kA1 + 4);
    v16h a;
    a[0]  = (_Float16)a0.x; a[1]  = (_Float16)a0.y; a[2]  = (_Float16)a0.z; a[3]  = (_Float16)a0.w;
    a[4]  = (_Float16)a1.x; a[5]  = (_Float16)a1.y; a[6]  = (_Float16)a1.z; a[7]  = (_Float16)a1.w;
    a[8]  = (_Float16)a2.x; a[9]  = (_Float16)a2.y; a[10] = (_Float16)a2.z; a[11] = (_Float16)a2.w;
    a[12] = (_Float16)a3.x; a[13] = (_Float16)a3.y; a[14] = (_Float16)a3.z; a[15] = (_Float16)a3.w;
    const v16h b = bp[kk * 32];
    acc = __builtin_amdgcn_wmma_f32_16x16x32_f16(false, a, false, b,
                                                 (short)0, acc, false, false);
  }

  const int mbase = tm * 16 + 8 * hw;
  float* __restrict__ cp = C + (size_t)mbase * NN + tn * 16 + r;
  if (mbase + 7 < M) {
#pragma unroll
    for (int v = 0; v < 8; ++v) cp[v * NN] = acc[v];
  } else {
#pragma unroll
    for (int v = 0; v < 8; ++v)
      if (mbase + v < M) cp[v * NN] = acc[v];
  }
}

// ---------------------------------------------------------------------------
// Per-node attention coefficients: as[n,h] = <xw[n,h,:], att_src[h,:]>, same ad
// ---------------------------------------------------------------------------
__global__ void gat_node_alpha(const float* __restrict__ xw,
                               const float* __restrict__ att_src,
                               const float* __restrict__ att_dst,
                               float* __restrict__ as_, float* __restrict__ ad_,
                               int nNodes, int H, int C)
{
  const int i = blockIdx.x * blockDim.x + threadIdx.x;
  if (i >= nNodes * H) return;
  const int n = i / H, h = i - n * H;
  const float4* __restrict__ row = (const float4*)(xw + (size_t)n * H * C + h * C);
  const float4* __restrict__ a1  = (const float4*)(att_src + h * C);
  const float4* __restrict__ a2  = (const float4*)(att_dst + h * C);
  float s1 = 0.f, s2 = 0.f;
  for (int c = 0; c < C / 4; ++c) {
    const float4 v = row[c], w1 = a1[c], w2 = a2[c];
    s1 = fmaf(v.x, w1.x, fmaf(v.y, w1.y, fmaf(v.z, w1.z, fmaf(v.w, w1.w, s1))));
    s2 = fmaf(v.x, w2.x, fmaf(v.y, w2.y, fmaf(v.z, w2.z, fmaf(v.w, w2.w, s2))));
  }
  as_[i] = s1; ad_[i] = s2;
}

// float <-> order-preserving unsigned key (atomic max over signed floats)
__device__ __forceinline__ unsigned fkey(float f) {
  int b = __float_as_int(f);
  return (unsigned)b ^ (((unsigned)(b >> 31)) | 0x80000000u);
}
__device__ __forceinline__ float fkey_inv(unsigned u) {
  int b = (u & 0x80000000u) ? (int)(u ^ 0x80000000u) : (int)~u;
  return __int_as_float(b);
}

// pass 1: e = leakyrelu(as[src]+ad[dst]); segment max over dst via atomicMax
__global__ void gat_edge_pass1(const int* __restrict__ ei, int E, int nNodes,
                               const float* __restrict__ as_, const float* __restrict__ ad_,
                               float* __restrict__ e_out, unsigned* __restrict__ mkeys, int H)
{
  const int i = blockIdx.x * blockDim.x + threadIdx.x;
  if (i >= (E + nNodes) * H) return;
  const int eid = i / H, h = i - eid * H;
  int s, d;
  if (eid < E) { s = ei[eid]; d = ei[E + eid]; } else { s = d = eid - E; }
  float e = as_[s * H + h] + ad_[d * H + h];
  e = (e > 0.f) ? e : e * NEG_SLOPE;
  e_out[i] = e;
  atomicMax(&mkeys[d * H + h], fkey(e));
}

// pass 2: ex = exp(e - m[dst]); segment sum over dst via atomicAdd
__global__ void gat_edge_pass2(const int* __restrict__ ei, int E, int nNodes,
                               const unsigned* __restrict__ mkeys,
                               float* __restrict__ e_inout, float* __restrict__ ssum, int H)
{
  const int i = blockIdx.x * blockDim.x + threadIdx.x;
  if (i >= (E + nNodes) * H) return;
  const int eid = i / H, h = i - eid * H;
  const int d = (eid < E) ? ei[E + eid] : (eid - E);
  const float ex = __expf(e_inout[i] - fkey_inv(mkeys[d * H + h]));
  e_inout[i] = ex;
  atomicAdd(&ssum[d * H + h], ex);
}

// pass 3: out[dst, f] += xw[src, f] * (ex/s[dst]); one thread per (edge, 4 chans)
__global__ void gat_edge_pass3(const int* __restrict__ ei, int E, int nNodes,
                               const float* __restrict__ xw, const float* __restrict__ ex,
                               const float* __restrict__ ssum, float* __restrict__ out,
                               int H, int C)
{
  const int F = H * C, G = F / 4;
  const int i = blockIdx.x * blockDim.x + threadIdx.x;
  if (i >= (E + nNodes) * G) return;
  const int eid = i / G, g = i - eid * G;
  const int f = g * 4, h = f / C;
  int s, d;
  if (eid < E) { s = ei[eid]; d = ei[E + eid]; } else { s = d = eid - E; }
  const float alpha = ex[eid * H + h] / ssum[d * H + h];
  const float4 v = *(const float4*)(xw + (size_t)s * F + f);
  float* __restrict__ o = out + (size_t)d * F + f;
  atomicAdd(o + 0, v.x * alpha);
  atomicAdd(o + 1, v.y * alpha);
  atomicAdd(o + 2, v.z * alpha);
  atomicAdd(o + 3, v.w * alpha);
}

__global__ void gat_bias_act(float* __restrict__ x, const float* __restrict__ b,
                             int nNodes, int F, int do_relu)
{
  const int i = blockIdx.x * blockDim.x + threadIdx.x;
  if (i >= nNodes * F) return;
  float v = x[i] + b[i % F];
  x[i] = do_relu ? fmaxf(v, 0.f) : v;
}

// score = [z[src]; z[dst]] . link_W + link_b
__global__ void gat_decode(const int* __restrict__ ei, int P,
                           const float* __restrict__ z, const float* __restrict__ lw,
                           const float* __restrict__ lb, float* __restrict__ out, int C)
{
  const int i = blockIdx.x * blockDim.x + threadIdx.x;
  if (i >= P) return;
  const float4* __restrict__ zs = (const float4*)(z + (size_t)ei[i] * C);
  const float4* __restrict__ zd = (const float4*)(z + (size_t)ei[P + i] * C);
  const float4* __restrict__ w0 = (const float4*)lw;
  const float4* __restrict__ w1 = (const float4*)(lw + C);
  float acc = lb[0];
  for (int c = 0; c < C / 4; ++c) {
    const float4 a = zs[c], b = zd[c], u = w0[c], v = w1[c];
    acc = fmaf(a.x, u.x, fmaf(a.y, u.y, fmaf(a.z, u.z, fmaf(a.w, u.w, acc))));
    acc = fmaf(b.x, v.x, fmaf(b.y, v.y, fmaf(b.z, v.z, fmaf(b.w, v.w, acc))));
  }
  out[i] = acc;
}

// ---------------------------------------------------------------------------
extern "C" void kernel_launch(void* const* d_in, const int* in_sizes, int n_in,
                              void* d_out, int out_size, void* d_ws, size_t ws_size,
                              hipStream_t stream)
{
  const int IN_C = 128, HIDC = 64, HEADS = 2;
  const int nN = in_sizes[0] / IN_C;     // 100000
  const int E  = in_sizes[1] / 2;        // 1600000
  const int P  = in_sizes[3] / 2;        // 262144
  const int E2 = E + nN;                 // with self-loops

  const float* x      = (const float*)d_in[0];
  const int*   ei     = (const int*)  d_in[1];
  const int*   pos_ei = (const int*)  d_in[3];
  const int*   neg_ei = (const int*)  d_in[4];
  const float* W1     = (const float*)d_in[5];
  const float* att_s1 = (const float*)d_in[6];
  const float* att_d1 = (const float*)d_in[7];
  const float* b1     = (const float*)d_in[8];
  const float* W2     = (const float*)d_in[9];
  const float* att_s2 = (const float*)d_in[10];
  const float* att_d2 = (const float*)d_in[11];
  const float* b2     = (const float*)d_in[12];
  const float* lw     = (const float*)d_in[13];
  const float* lb     = (const float*)d_in[14];
  float* out = (float*)d_out;

  // workspace layout (packed B first for 32B alignment)
  v16h* Bp1 = (v16h*)d_ws;                          // (128/16)*(128/32)*32 = 1024
  v16h* Bp2 = Bp1 + 1024;                           // (64/16)*(128/32)*32  = 512
  float* xw1 = (float*)(Bp2 + 512);                 // N*128
  float* h   = xw1 + (size_t)nN * 128;              // N*128
  float* xw2 = h   + (size_t)nN * 128;              // N*64
  float* z   = xw2 + (size_t)nN * 64;               // N*64
  float* as1 = z   + (size_t)nN * 64;               // N*2
  float* ad1 = as1 + (size_t)nN * 2;                // N*2
  unsigned* m1 = (unsigned*)(ad1 + (size_t)nN * 2); // N*2
  float* s1  = (float*)(m1 + (size_t)nN * 2);       // N*2
  float* as2 = s1  + (size_t)nN * 2;                // N
  float* ad2 = as2 + nN;                            // N
  unsigned* m2 = (unsigned*)(ad2 + nN);             // N
  float* s2  = (float*)(m2 + nN);                   // N
  float* e1  = s2 + nN;                             // E2*2
  float* e2  = e1 + (size_t)E2 * 2;                 // E2
  const size_t need = (size_t)((char*)(e2 + E2) - (char*)d_ws);
  if (need > ws_size) return;                       // insufficient scratch

  const int TB = 256;
  auto cdiv = [](long a, long b) { return (unsigned)((a + b - 1) / b); };

  // ---- layer 1: xw1 = x @ W1 (WMMA) ----
  gat_pack_b<<<cdiv(8 * 4 * 32, TB), TB, 0, stream>>>(W1, Bp1, 128, 128);
  {
    const int tiles = ((nN + 15) / 16) * (128 / 16);
    gat_wmma_gemm<128, 128><<<cdiv(tiles, 8), 256, 0, stream>>>(x, Bp1, xw1, nN);
  }
  gat_node_alpha<<<cdiv((long)nN * HEADS, TB), TB, 0, stream>>>(
      xw1, att_s1, att_d1, as1, ad1, nN, HEADS, HIDC);
  hipMemsetAsync(m1, 0, (size_t)nN * 2 * sizeof(unsigned), stream);
  hipMemsetAsync(s1, 0, (size_t)nN * 2 * sizeof(float), stream);
  hipMemsetAsync(h,  0, (size_t)nN * 128 * sizeof(float), stream);
  gat_edge_pass1<<<cdiv((long)E2 * HEADS, TB), TB, 0, stream>>>(
      ei, E, nN, as1, ad1, e1, m1, HEADS);
  gat_edge_pass2<<<cdiv((long)E2 * HEADS, TB), TB, 0, stream>>>(
      ei, E, nN, m1, e1, s1, HEADS);
  gat_edge_pass3<<<cdiv((long)E2 * 32, TB), TB, 0, stream>>>(
      ei, E, nN, xw1, e1, s1, h, HEADS, HIDC);
  gat_bias_act<<<cdiv((long)nN * 128, TB), TB, 0, stream>>>(h, b1, nN, 128, 1);

  // ---- layer 2: xw2 = h @ W2 (WMMA) ----
  gat_pack_b<<<cdiv(4 * 4 * 32, TB), TB, 0, stream>>>(W2, Bp2, 64, 128);
  {
    const int tiles = ((nN + 15) / 16) * (64 / 16);
    gat_wmma_gemm<128, 64><<<cdiv(tiles, 8), 256, 0, stream>>>(h, Bp2, xw2, nN);
  }
  gat_node_alpha<<<cdiv(nN, TB), TB, 0, stream>>>(
      xw2, att_s2, att_d2, as2, ad2, nN, 1, HIDC);
  hipMemsetAsync(m2, 0, (size_t)nN * sizeof(unsigned), stream);
  hipMemsetAsync(s2, 0, (size_t)nN * sizeof(float), stream);
  hipMemsetAsync(z,  0, (size_t)nN * 64 * sizeof(float), stream);
  gat_edge_pass1<<<cdiv(E2, TB), TB, 0, stream>>>(ei, E, nN, as2, ad2, e2, m2, 1);
  gat_edge_pass2<<<cdiv(E2, TB), TB, 0, stream>>>(ei, E, nN, m2, e2, s2, 1);
  gat_edge_pass3<<<cdiv((long)E2 * 16, TB), TB, 0, stream>>>(
      ei, E, nN, xw2, e2, s2, z, 1, HIDC);
  gat_bias_act<<<cdiv((long)nN * 64, TB), TB, 0, stream>>>(z, b2, nN, 64, 0);

  // ---- link decode: pos then neg, concatenated in d_out ----
  gat_decode<<<cdiv(P, TB), TB, 0, stream>>>(pos_ei, P, z, lw, lb, out, HIDC);
  gat_decode<<<cdiv(P, TB), TB, 0, stream>>>(neg_ei, P, z, lw, lb, out + P, HIDC);
}
